// MLASelfAttentionWithMoBA_20100446945834
// MI455X (gfx1250) — matrix operand order repaired and checked by reading, not compiled, not run on hardware
//
#include <hip/hip_runtime.h>

// ---------------- types ----------------
typedef __bf16 bf16_t;
typedef __attribute__((ext_vector_type(8)))  __bf16 bf16x8;
typedef __attribute__((ext_vector_type(16))) __bf16 bf16x16;
typedef __attribute__((ext_vector_type(8)))  float  f32x8;
typedef __attribute__((ext_vector_type(4)))  float  f32x4;
typedef __attribute__((ext_vector_type(4)))  int    i32x4;

// CDNA5 async global->LDS copy path (ASYNCcnt-tracked, bypasses VGPRs).
// Probe round 2: builtin exists with signature (i32x4 AS1*, i32x4 AS3*, imm, imm).
#if defined(__has_builtin)
#if __has_builtin(__builtin_amdgcn_global_load_async_to_lds_b128)
#define HAVE_ASYNC_LDS 1
#endif
#endif

__device__ __forceinline__ void async_copy16(const bf16_t* g, bf16_t* l) {
#ifdef HAVE_ASYNC_LDS
    __builtin_amdgcn_global_load_async_to_lds_b128(
        (__attribute__((address_space(1))) i32x4*)(
            (__attribute__((address_space(1))) void*)(bf16_t*)g),
        (__attribute__((address_space(3))) i32x4*)(
            (__attribute__((address_space(3))) void*)l),
        0, 0);
#else
    *(bf16x8*)l = *(const bf16x8*)g;
#endif
}

__device__ __forceinline__ void async_wait0() {
#ifdef HAVE_ASYNC_LDS
#if __has_builtin(__builtin_amdgcn_s_wait_asynccnt)
    __builtin_amdgcn_s_wait_asynccnt(0);
#else
    asm volatile("s_wait_asynccnt 0" ::: "memory");
#endif
#endif
}

// Assemble a 16x32 bf16 A-fragment from two 8-element contiguous chunks
// (each one ds_load_b128 / global_load_b128), per the CDNA5 16-bit A layout.
__device__ __forceinline__ bf16x16 frag2(const bf16_t* lo_p, const bf16_t* hi_p) {
    bf16x8 lo = *(const bf16x8*)lo_p;
    bf16x8 hi = *(const bf16x8*)hi_p;
    return __builtin_shufflevector(lo, hi, 0,1,2,3,4,5,6,7,8,9,10,11,12,13,14,15);
}
// B-fragment: two 8-element k-chunks at p and p+16 (same k order as the A layout).
__device__ __forceinline__ bf16x16 frag16(const bf16_t* p) { return frag2(p, p + 16); }

__device__ __forceinline__ f32x8 wmma_bf16(bf16x16 a, bf16x16 b, f32x8 c) {
    return __builtin_amdgcn_wmma_f32_16x16x32_bf16(false, a, false, b, (short)0, c, false, false);
}

// ---------------- fp32 -> bf16 convert (vectorized, 8 elem/thread) ----------------
__global__ __launch_bounds__(256) void cvt_bf16(const float* __restrict__ in,
                                                bf16_t* __restrict__ out, int n8) {
    int i = blockIdx.x * 256 + threadIdx.x;
    if (i < n8) {
        f32x4 a = *(const f32x4*)&in[i * 8];
        f32x4 b = *(const f32x4*)&in[i * 8 + 4];
        bf16x8 o;
        #pragma unroll
        for (int k = 0; k < 4; ++k) { o[k] = (bf16_t)a[k]; o[k + 4] = (bf16_t)b[k]; }
        *(bf16x8*)&out[i * 8] = o;
    }
}

// ---------------- tiled bf16 WMMA GEMM: C[M,N] = A[M,K] * B[K,N] ----------------
// Block 256 threads = 8 waves. Tile BM=128, BN=128, BK=32, double-buffered LDS.
// A tile staged via async global->LDS; B tile loaded to regs early, transposed
// into LDS after the current tile's WMMAs (latency hidden under compute).
__global__ __launch_bounds__(256) void gemm_bf16(const bf16_t* __restrict__ A,
                                                 const bf16_t* __restrict__ B,
                                                 float* __restrict__ C,
                                                 int M, int N, int K) {
    __shared__ __attribute__((aligned(16))) bf16_t As[2][128 * 32];
    __shared__ __attribute__((aligned(16))) bf16_t Bs[2][128 * 32]; // Bs[n][k]
    const int tid  = threadIdx.x;
    const int nb   = N >> 7;
    const int bm   = blockIdx.x / nb, bn = blockIdx.x % nb;
    const int m0   = bm * 128, n0 = bn * 128;
    const int w    = tid >> 5, lane = tid & 31;
    const int hf   = lane >> 4, ln = lane & 15;
    const int wm   = w & 3, wn = w >> 2;

    f32x8 acc[2][4] = {};

    // ---- prologue: stage tile 0 ----
    {
        #pragma unroll
        for (int it = 0; it < 2; ++it) {
            int e = tid * 8 + it * 2048;
            int r = e >> 5, c = e & 31;
            async_copy16(&A[(size_t)(m0 + r) * K + c], &As[0][e]);
        }
        bf16x8 bv[2];
        #pragma unroll
        for (int it = 0; it < 2; ++it) {
            int e = tid * 8 + it * 2048;
            int kr = e >> 7, nc = e & 127;
            bv[it] = *(const bf16x8*)&B[(size_t)kr * N + n0 + nc];
        }
        #pragma unroll
        for (int it = 0; it < 2; ++it) {
            int e = tid * 8 + it * 2048;
            int kr = e >> 7, nc = e & 127;
            #pragma unroll
            for (int i = 0; i < 8; ++i) Bs[0][(nc + i) * 32 + kr] = bv[it][i];
        }
    }
    async_wait0();
    __syncthreads();

    for (int kb = 0; kb < K; kb += 32) {
        const int cur = (kb >> 5) & 1, nxt = cur ^ 1;
        const bool pf = (kb + 32) < K;
        bf16x8 bv[2];
        if (pf) {
            // prefetch next tile: async A -> LDS, B -> registers
            #pragma unroll
            for (int it = 0; it < 2; ++it) {
                int e = tid * 8 + it * 2048;
                int r = e >> 5, c = e & 31;
                async_copy16(&A[(size_t)(m0 + r) * K + kb + 32 + c], &As[nxt][e]);
            }
            #pragma unroll
            for (int it = 0; it < 2; ++it) {
                int e = tid * 8 + it * 2048;
                int kr = e >> 7, nc = e & 127;
                bv[it] = *(const bf16x8*)&B[(size_t)(kb + 32 + kr) * N + n0 + nc];
            }
        }

        // ---- compute on current buffers ----
        bf16x16 af[2];
        #pragma unroll
        for (int mi = 0; mi < 2; ++mi) {
            const bf16_t* p = &As[cur][(wm * 32 + mi * 16 + ln) * 32 + hf * 8];
            af[mi] = frag2(p, p + 16);
        }
        #pragma unroll
        for (int ni = 0; ni < 4; ++ni) {
            bf16x16 bfv = frag16(&Bs[cur][(wn * 64 + ni * 16 + ln) * 32 + hf * 8]);
            #pragma unroll
            for (int mi = 0; mi < 2; ++mi)
                acc[mi][ni] = wmma_bf16(af[mi], bfv, acc[mi][ni]);
        }

        if (pf) { // transpose-scatter B into the next buffer after the math
            #pragma unroll
            for (int it = 0; it < 2; ++it) {
                int e = tid * 8 + it * 2048;
                int kr = e >> 7, nc = e & 127;
                #pragma unroll
                for (int i = 0; i < 8; ++i) Bs[nxt][(nc + i) * 32 + kr] = bv[it][i];
            }
        }
        async_wait0();
        __syncthreads();
    }

    #pragma unroll
    for (int mi = 0; mi < 2; ++mi)
        #pragma unroll
        for (int ni = 0; ni < 4; ++ni)
            #pragma unroll
            for (int r = 0; r < 8; ++r)
                C[(size_t)(m0 + wm * 32 + mi * 16 + hf * 8 + r) * N +
                  n0 + wn * 64 + ni * 16 + ln] = acc[mi][ni][r];
}

// ---------------- up-projection + partial interleaved RoPE ----------------
// One thread per (b,t,h,d). Emits bf16 q,k in [B,H,T,64] and v transposed [B,H,64,T].
__global__ __launch_bounds__(256) void upproj_rope(
        const float* __restrict__ qlat, const float* __restrict__ klat,
        const float* __restrict__ vlat,
        const float* __restrict__ Wqc, const float* __restrict__ Wqe,
        const float* __restrict__ Wkc, const float* __restrict__ Wke,
        const float* __restrict__ Wv,
        bf16_t* __restrict__ qws, bf16_t* __restrict__ kws, bf16_t* __restrict__ vt) {
    int g = blockIdx.x * 256 + threadIdx.x;
    int d = g & 63, h = (g >> 6) & 15, t = (g >> 10) & 1023, b = g >> 20;
    size_t row = (size_t)(b * 1024 + t) * 1024 + h * 64;
    const float* ql = qlat + row;
    const float* kl = klat + row;
    const float* vl = vlat + row;

    // v
    float vv = 0.f;
    #pragma unroll 8
    for (int l = 0; l < 64; ++l) vv += vl[l] * Wv[l * 64 + d];
    vt[((size_t)(b * 16 + h) * 64 + d) * 1024 + t] = (bf16_t)vv;

    float sn = 0.f, cs = 1.f, rotsign = 0.f;
    int j = d - 32;
    if (d >= 32) {
        float invf = __powf(10000.0f, -(float)(j & 15) / 16.0f);
        __sincosf((float)t * invf, &sn, &cs);
        rotsign = (j & 1) ? 1.0f : -1.0f;
    }
    // q
    float qv;
    if (d < 32) {
        qv = 0.f;
        #pragma unroll 8
        for (int l = 0; l < 64; ++l) qv += ql[l] * Wqc[l * 32 + d];
    } else {
        float e0 = 0.f, e1 = 0.f;
        #pragma unroll 8
        for (int l = 0; l < 64; ++l) {
            float x = ql[l];
            e0 += x * Wqe[l * 32 + j];
            e1 += x * Wqe[l * 32 + (j ^ 1)];
        }
        qv = e0 * cs + rotsign * e1 * sn;
    }
    qws[((size_t)(b * 16 + h) * 1024 + t) * 64 + d] = (bf16_t)qv;
    // k
    float kv;
    if (d < 32) {
        kv = 0.f;
        #pragma unroll 8
        for (int l = 0; l < 64; ++l) kv += kl[l] * Wkc[l * 32 + d];
    } else {
        float e0 = 0.f, e1 = 0.f;
        #pragma unroll 8
        for (int l = 0; l < 64; ++l) {
            float x = kl[l];
            e0 += x * Wke[l * 32 + j];
            e1 += x * Wke[l * 32 + (j ^ 1)];
        }
        kv = e0 * cs + rotsign * e1 * sn;
    }
    kws[((size_t)(b * 16 + h) * 1024 + t) * 64 + d] = (bf16_t)kv;
}

// ---------------- flash-style causal attention ----------------
// Grid: 512 blocks = (b, h, qtile of 128 rows). 8 waves, wave owns 16 q-rows.
// Q,K in [B,H,T,64] bf16; V transposed [B,H,64,T] bf16. Output bf16 [B,T,H*64].
__global__ __launch_bounds__(256) void attn_flash(
        const bf16_t* __restrict__ Q, const bf16_t* __restrict__ Km,
        const bf16_t* __restrict__ Vt, bf16_t* __restrict__ Y) {
    __shared__ __attribute__((aligned(16))) bf16_t Ps[8 * 16 * 64]; // per-wave 16x64 P tile
    const int tid = threadIdx.x;
    const int w = tid >> 5, lane = tid & 31, hf = lane >> 4, ln = lane & 15;
    const int qt = blockIdx.x & 7;
    const int h  = (blockIdx.x >> 3) & 15;
    const int b  = blockIdx.x >> 7;
    const int bh = b * 16 + h;
    const int qrow0 = qt * 128 + w * 16;

    // Preload Q A-fragments (two K=32 steps over d) straight from global.
    bf16x16 aq[2];
    const bf16_t* qbase = Q + ((size_t)bh * 1024 + qrow0 + ln) * 64;
    #pragma unroll
    for (int s = 0; s < 2; ++s) {
        const bf16_t* p = qbase + s * 32 + hf * 8;
        aq[s] = frag2(p, p + 16);
    }

    f32x8 acc[4] = {};
    float m_i[8], l_i[8];
    #pragma unroll
    for (int r = 0; r < 8; ++r) { m_i[r] = -3.0e38f; l_i[r] = 0.f; }
    bf16_t* Pw = &Ps[w * 16 * 64];

    const int nkb = qt * 2 + 2; // causal: only key blocks covering keys <= last q row
    for (int kb = 0; kb < nkb; ++kb) {
        const int key0 = kb * 64;
        // S = Q * K^T  (K rows are k-contiguous => direct B-fragments)
        f32x8 sacc[4] = {};
        #pragma unroll
        for (int j = 0; j < 4; ++j) {
            const bf16_t* kp = Km + ((size_t)bh * 1024 + key0 + j * 16 + ln) * 64;
            #pragma unroll
            for (int s = 0; s < 2; ++s)
                sacc[j] = wmma_bf16(aq[s], frag16(kp + s * 32 + hf * 8), sacc[j]);
        }
        // scale + causal mask + row max (rows live at m = hf*8+r, cols at n = ln+16j)
        float mx[8];
        #pragma unroll
        for (int r = 0; r < 8; ++r) mx[r] = -3.0e38f;
        #pragma unroll
        for (int j = 0; j < 4; ++j)
            #pragma unroll
            for (int r = 0; r < 8; ++r) {
                float v = sacc[j][r] * 0.125f; // 1/sqrt(64)
                int qg = qrow0 + hf * 8 + r;
                int kg = key0 + j * 16 + ln;
                if (kg > qg) v = -1.0e30f;
                sacc[j][r] = v;
                mx[r] = fmaxf(mx[r], v);
            }
        #pragma unroll
        for (int r = 0; r < 8; ++r)
            #pragma unroll
            for (int off = 1; off < 16; off <<= 1)
                mx[r] = fmaxf(mx[r], __shfl_xor(mx[r], off, 32));
        float corr[8], rs[8];
        #pragma unroll
        for (int r = 0; r < 8; ++r) {
            float nm = fmaxf(m_i[r], mx[r]);
            corr[r] = __expf(m_i[r] - nm);
            m_i[r] = nm;
            rs[r] = 0.f;
        }
        // P = exp(S - m); spill to per-wave LDS in [m][k] layout for re-fragmenting
        #pragma unroll
        for (int j = 0; j < 4; ++j)
            #pragma unroll
            for (int r = 0; r < 8; ++r) {
                float p = __expf(sacc[j][r] - m_i[r]);
                rs[r] += p;
                Pw[(hf * 8 + r) * 64 + j * 16 + ln] = (bf16_t)p;
            }
        #pragma unroll
        for (int r = 0; r < 8; ++r) {
            #pragma unroll
            for (int off = 1; off < 16; off <<= 1) rs[r] += __shfl_xor(rs[r], off, 32);
            l_i[r] = l_i[r] * corr[r] + rs[r];
        }
        #pragma unroll
        for (int j = 0; j < 4; ++j)
            #pragma unroll
            for (int r = 0; r < 8; ++r) acc[j][r] *= corr[r];

        // DS ops are in-order per wave; fence compiler + wait for stores.
        asm volatile("s_wait_dscnt 0" ::: "memory");

        bf16x16 ap[2];
        #pragma unroll
        for (int s = 0; s < 2; ++s) {
            const bf16_t* p = &Pw[ln * 64 + s * 32 + hf * 8];
            ap[s] = frag2(p, p + 16);
        }
        // O += P * V  (V^T is k-contiguous => direct B-fragments from global)
        #pragma unroll
        for (int j = 0; j < 4; ++j) {
            const bf16_t* vp = Vt + ((size_t)bh * 64 + j * 16 + ln) * 1024 + key0;
            #pragma unroll
            for (int s = 0; s < 2; ++s)
                acc[j] = wmma_bf16(ap[s], frag16(vp + s * 32 + hf * 8), acc[j]);
        }
    }
    // write back un-transposed: [B,T,H*64] bf16 for the Wc GEMM
    #pragma unroll
    for (int j = 0; j < 4; ++j)
        #pragma unroll
        for (int r = 0; r < 8; ++r) {
            float y = acc[j][r] / l_i[r];
            Y[((size_t)b * 1024 + qrow0 + hf * 8 + r) * 1024 + h * 64 + j * 16 + ln] =
                (bf16_t)y;
        }
}

// ---------------- host-side launcher ----------------
extern "C" void kernel_launch(void* const* d_in, const int* in_sizes, int n_in,
                              void* d_out, int out_size, void* d_ws, size_t ws_size,
                              hipStream_t stream) {
    (void)in_sizes; (void)n_in; (void)out_size; (void)ws_size;
    const float* x       = (const float*)d_in[0];
    const float* Wq_down = (const float*)d_in[1];
    const float* Wk_down = (const float*)d_in[2];
    const float* Wv_down = (const float*)d_in[3];
    const float* Wq_up_c = (const float*)d_in[4];
    const float* Wq_up_e = (const float*)d_in[5];
    const float* Wk_up_c = (const float*)d_in[6];
    const float* Wk_up_e = (const float*)d_in[7];
    const float* Wv_up   = (const float*)d_in[8];
    const float* Wc      = (const float*)d_in[9];
    float* out = (float*)d_out;

    char* ws = (char*)d_ws;
    const size_t MB = 1u << 20;
    bf16_t* xbf  = (bf16_t*)(ws);             // 8 MB  (later aliased by ybf)
    bf16_t* ybf  = xbf;                       // alias: x dead after down-proj GEMMs
    bf16_t* wqbf = (bf16_t*)(ws + 8  * MB);   // 2 MB each
    bf16_t* wkbf = (bf16_t*)(ws + 10 * MB);
    bf16_t* wvbf = (bf16_t*)(ws + 12 * MB);
    bf16_t* wcbf = (bf16_t*)(ws + 14 * MB);
    float*  qlat = (float*) (ws + 16 * MB);   // 16 MB fp32
    bf16_t* qws  = (bf16_t*)(ws + 32 * MB);   // 8 MB  [B,H,T,64]
    bf16_t* kws  = (bf16_t*)(ws + 40 * MB);   // 8 MB  [B,H,T,64]
    bf16_t* vt   = (bf16_t*)(ws + 48 * MB);   // 8 MB  [B,H,64,T]

    float* klat_out = out + 4194304; // k_lat output region [4,1024,16,64]
    float* vlat_out = out + 8388608; // v_lat output region

    const int M = 4096, N = 1024, K = 1024;

    cvt_bf16<<<(M * K) / 2048, 256, 0, stream>>>(x, xbf, (M * K) / 8);
    cvt_bf16<<<(K * N) / 2048, 256, 0, stream>>>(Wq_down, wqbf, (K * N) / 8);
    cvt_bf16<<<(K * N) / 2048, 256, 0, stream>>>(Wk_down, wkbf, (K * N) / 8);
    cvt_bf16<<<(K * N) / 2048, 256, 0, stream>>>(Wv_down, wvbf, (K * N) / 8);
    cvt_bf16<<<(K * N) / 2048, 256, 0, stream>>>(Wc,      wcbf, (K * N) / 8);

    dim3 gg((M / 128) * (N / 128));
    gemm_bf16<<<gg, 256, 0, stream>>>(xbf, wqbf, qlat,     M, N, K);
    gemm_bf16<<<gg, 256, 0, stream>>>(xbf, wkbf, klat_out, M, N, K);
    gemm_bf16<<<gg, 256, 0, stream>>>(xbf, wvbf, vlat_out, M, N, K);

    upproj_rope<<<4194304 / 256, 256, 0, stream>>>(
        qlat, klat_out, vlat_out, Wq_up_c, Wq_up_e, Wk_up_c, Wk_up_e, Wv_up,
        qws, kws, vt);

    attn_flash<<<512, 256, 0, stream>>>(qws, kws, vt, ybf);

    gemm_bf16<<<gg, 256, 0, stream>>>(ybf, wcbf, out, M, N, K);
}
